// Model_80367428042972
// MI455X (gfx1250) — matrix operand order repaired and checked by reading, not compiled, not run on hardware
//
#include <hip/hip_runtime.h>
#include <hip/hip_bf16.h>
#include <math.h>

// ---------------------------------------------------------------------------
// MI455X (gfx1250) implementation.
// Precision: bf16 WMMA (v_wmma_f32_16x16x32_bf16) with f32 accumulate.
// Roofline: ~110 GFLOP vs ~250MB touched -> balanced at bf16; the whole 92MB
// activation tensor is L2-resident (192MB), so bf16 halves HBM traffic and
// rides the matrix pipe instead of the f32 VALU.
// This revision forces the GEMM A/B/C pointers into address space 1 (global)
// so the B-fragment loads emit global_load_b128 instead of flat_load_b128
// (flat ties up DScnt + the LDS arbitration path on CDNA5).
// ---------------------------------------------------------------------------

typedef __bf16 bf16_t;
typedef __attribute__((ext_vector_type(16))) __bf16 v16bf;
typedef __attribute__((ext_vector_type(8)))  __bf16 v8bf;
typedef __attribute__((ext_vector_type(8)))  float  v8f;

// Global (addrspace 1) pointer flavors: C-style casts below perform the
// addrspacecast (the only cast form clang accepts for this).
typedef const __bf16 __attribute__((address_space(1)))* gcbf_p;
typedef const v16bf  __attribute__((address_space(1)))* gv16_p;
typedef const v8bf   __attribute__((address_space(1)))* gv8_p;
typedef float        __attribute__((address_space(1)))* gf_p;
typedef __bf16       __attribute__((address_space(1)))* gbf_p;

#define B_     32
#define CIN    64
#define SEQ    512
#define PRED   192
#define OUTL   704        // SEQ + PRED
#define D_     128
#define PL_    16
#define PNUM_  88
#define POLD   64
#define PPAD   96         // PNUM_ padded to multiple of 32 for K of chan GEMMs
#define NROWS  2048       // B_ * CIN
#define KHEAD  (D_*PPAD)  // 12288
#define MAXEL  25165824ll // max(NROWS*PNUM_*D_, D_*NROWS*PPAD)

__device__ __forceinline__ bf16_t to_bf(float f) { return (bf16_t)f; }
__device__ __forceinline__ float gelu_exact(float x) {
  return 0.5f * x * (1.0f + erff(x * 0.70710678118654752f));
}

// ---------------------------------------------------------------------------
// Branch-free GEMM tile core, specialized on NT (number of 16-wide N tiles).
// C = A (MxK row-major, stride lda) x Bw^T (Bw is [N][K] row-major) + bias.
// One 256-thread block = 8 wave32; wave w owns rows [w*16, w*16+16).
// ---------------------------------------------------------------------------
template<int EPI, int NT>
__device__ __forceinline__ void gemm_tile(
    const bf16_t* __restrict__ A, int lda,
    const bf16_t* __restrict__ Bw,
    const float*  __restrict__ bias, int biasN,
    float* __restrict__ Cf, int ldc,
    bf16_t* __restrict__ Cb, int ldcb,
    int K, int mBase, int nBase)
{
  const int wave = threadIdx.x >> 5;
  const int lane = threadIdx.x & 31;
  const int mloc = lane & 15;   // A row within tile / B,C column within tile
  const int half = lane >> 4;   // selects K sub-range (A/B) and M offset (C)

  // A fragment (16x32 bf16): lane(m,h) holds K = h*16+{0..15}
  gcbf_p a0 = (gcbf_p)(A + (long long)(mBase + wave * 16 + mloc) * lda + half * 16);
  gcbf_p a1 = a0 + 8;
  // B fragment (32x16 bf16): lane holds column n=mloc, 16 consecutive K at
  // k0 + half*16 -> one contiguous 32B chunk of Bw row n.
  gcbf_p bp[NT];
#pragma unroll
  for (int nt = 0; nt < NT; ++nt)
    bp[nt] = (gcbf_p)(Bw + (long long)(nBase + nt * 16 + mloc) * K + half * 16);

  v8f acc[NT] = {};
  for (int k0 = 0; k0 < K; k0 += 32) {
    union { v16bf v; v8bf h[2]; } af;
    af.h[0] = *(gv8_p)a0;
    af.h[1] = *(gv8_p)a1;
    v16bf bv[NT];
#pragma unroll
    for (int nt = 0; nt < NT; ++nt)
      bv[nt] = *(gv16_p)bp[nt];
    if (k0 + 64 <= K)
      __builtin_prefetch((const void*)(A + (long long)(mBase + wave * 16 + mloc) * lda + k0 + 32), 0, 3);
#pragma unroll
    for (int nt = 0; nt < NT; ++nt)
      acc[nt] = __builtin_amdgcn_wmma_f32_16x16x32_bf16(
          false, af.v, false, bv[nt], (short)0, acc[nt], false, false);
    a0 += 32; a1 += 32;
#pragma unroll
    for (int nt = 0; nt < NT; ++nt) bp[nt] += 32;
  }

  gf_p  cf = (gf_p)Cf;
  gbf_p cb = (gbf_p)Cb;
#pragma unroll
  for (int nt = 0; nt < NT; ++nt) {
    const int col = nBase + nt * 16 + mloc;
    const float bvs = (bias && col < biasN) ? bias[col] : 0.0f;
#pragma unroll
    for (int r = 0; r < 8; ++r) {
      const int row = mBase + wave * 16 + r + half * 8; // C: VGPR r -> M=r / r+8
      float v = acc[nt][r] + bvs;
      if (EPI == 1) v = gelu_exact(v);
      if (Cf) cf[(long long)row * ldc  + col] = v;
      if (Cb) cb[(long long)row * ldcb + col] = to_bf(v);
    }
  }
}

template<int EPI>
__global__ __launch_bounds__(256)
void gemm_bf16_wmma(const bf16_t* __restrict__ A, long long aBatch, int lda,
                    const bf16_t* __restrict__ Bw, long long bBatch,
                    const float*  __restrict__ bias, long long biasBatch, int biasN,
                    float* __restrict__ Cf, long long cfBatch, int ldc,
                    bf16_t* __restrict__ Cb, long long cbBatch, int ldcb,
                    int N, int K)
{
  const int batch = blockIdx.z;
  A  += (long long)batch * aBatch;
  Bw += (long long)batch * bBatch;
  if (bias) bias += (long long)batch * biasBatch;
  if (Cf)   Cf   += (long long)batch * cfBatch;
  if (Cb)   Cb   += (long long)batch * cbBatch;

  const int mBase = blockIdx.x * 128;
  const int nBase = blockIdx.y * 128;
  int nTiles = (N - nBase) >> 4;
  if (nTiles > 8) nTiles = 8;

  switch (nTiles) {
    case 8: gemm_tile<EPI, 8>(A, lda, Bw, bias, biasN, Cf, ldc, Cb, ldcb, K, mBase, nBase); break;
    case 6: gemm_tile<EPI, 6>(A, lda, Bw, bias, biasN, Cf, ldc, Cb, ldcb, K, mBase, nBase); break;
    case 4: gemm_tile<EPI, 4>(A, lda, Bw, bias, biasN, Cf, ldc, Cb, ldcb, K, mBase, nBase); break;
    default:
      for (int t = 0; t < nTiles; ++t)
        gemm_tile<EPI, 1>(A, lda, Bw, bias, biasN, Cf, ldc, Cb, ldcb, K, mBase, nBase + t * 16);
      break;
  }
}

// ---------------------------------------------------------------------------
// RevIN: per (b,c) mean/std over SEQ, write normalized row as bf16.
// ---------------------------------------------------------------------------
__global__ __launch_bounds__(256)
void revin_kernel(const float* __restrict__ x, const float* __restrict__ rw,
                  const float* __restrict__ rb, float* __restrict__ meanO,
                  float* __restrict__ stdO, bf16_t* __restrict__ xn)
{
  const int n = blockIdx.x;
  const int bb = n >> 6, c = n & 63;
  const float* base = x + (long long)bb * (SEQ * CIN) + c;
  __shared__ float sx[SEQ];
  float s = 0.f, q = 0.f;
  for (int t = threadIdx.x; t < SEQ; t += 256) {
    float v = base[(long long)t * CIN];
    sx[t] = v; s += v; q += v * v;
  }
#pragma unroll
  for (int off = 16; off > 0; off >>= 1) { s += __shfl_down(s, off, 32); q += __shfl_down(q, off, 32); }
  __shared__ float r1[8], r2[8];
  const int wid = threadIdx.x >> 5;
  if ((threadIdx.x & 31) == 0) { r1[wid] = s; r2[wid] = q; }
  __syncthreads();
  s = 0.f; q = 0.f;
#pragma unroll
  for (int i = 0; i < 8; ++i) { s += r1[i]; q += r2[i]; }
  const float mu = s * (1.f / SEQ);
  const float sd = sqrtf(q * (1.f / SEQ) - mu * mu + 1e-5f);
  if (threadIdx.x == 0) { meanO[n] = mu; stdO[n] = sd; }
  const float inv = 1.f / sd, wgt = rw[c], bia = rb[c];
  for (int t = threadIdx.x; t < SEQ; t += 256)
    xn[(long long)n * SEQ + t] = to_bf((sx[t] - mu) * inv * wgt + bia);
}

// ---------------------------------------------------------------------------
// Positional embedding table pe[PNUM_][D_]
// ---------------------------------------------------------------------------
__global__ void pe_init_kernel(float* __restrict__ pe)
{
  const int i = blockIdx.x * 256 + threadIdx.x;
  if (i >= PNUM_ * D_) return;
  const int p = i / D_, d = i % D_;
  const float div = expf(-(float)(d & ~1) * (logf(10000.f) / (float)D_));
  pe[i] = (d & 1) ? cosf((float)p * div) : sinf((float)p * div);
}

// ---------------------------------------------------------------------------
// Patch embed: out[n][p][d] = sum_k src[n, clamp(p*8+k)] * embW[d][k] + pe[p][d]
// src element (n,s) at (n/64)*sb + (n%64)*sc + s*ss  (handles both layouts)
// ---------------------------------------------------------------------------
__global__ __launch_bounds__(128)
void patch_embed_kernel(const float* __restrict__ src, long long sb, long long sc,
                        long long ss, int Lsrc, int P,
                        const float* __restrict__ embW, const float* __restrict__ pe,
                        float* __restrict__ outF, bf16_t* __restrict__ outB)
{
  const int n = blockIdx.x, p = blockIdx.y, d = threadIdx.x;
  __shared__ float patch[PL_];
  if (d < PL_) {
    int sidx = p * 8 + d;
    if (sidx > Lsrc - 1) sidx = Lsrc - 1; // edge pad
    patch[d] = src[(long long)(n >> 6) * sb + (long long)(n & 63) * sc + (long long)sidx * ss];
  }
  __syncthreads();
  float acc = pe[p * D_ + d];
#pragma unroll
  for (int k = 0; k < PL_; ++k) acc += patch[k] * embW[d * PL_ + k];
  const long long o = ((long long)n * P + p) * D_ + d;
  outF[o] = acc;
  if (outB) outB[o] = to_bf(acc);
}

// ---------------------------------------------------------------------------
// LayerNorm over D_=128 (one row per block). src viewed [R/Pv][Ps] rows, dst
// compact. Optional residual (compact layout) and optional bf16 mirror.
// ---------------------------------------------------------------------------
__global__ __launch_bounds__(128)
void ln_rows_kernel(const float* __restrict__ src, const float* __restrict__ res,
                    float* __restrict__ dst, bf16_t* __restrict__ dstb,
                    const float* __restrict__ w, const float* __restrict__ b,
                    int Ps, int Pv)
{
  const long long r = blockIdx.x;
  const int d = threadIdx.x;
  const long long so = ((r / Pv) * Ps + (r % Pv)) * (long long)D_ + d;
  float v = src[so];
  if (res) v += res[r * D_ + d];
  float s = v, q = v * v;
#pragma unroll
  for (int off = 16; off > 0; off >>= 1) { s += __shfl_down(s, off, 32); q += __shfl_down(q, off, 32); }
  __shared__ float s1[4], s2[4];
  const int wid = d >> 5;
  if ((d & 31) == 0) { s1[wid] = s; s2[wid] = q; }
  __syncthreads();
  s = s1[0] + s1[1] + s1[2] + s1[3];
  q = s2[0] + s2[1] + s2[2] + s2[3];
  const float mu  = s * (1.f / 128.f);
  const float var = q * (1.f / 128.f) - mu * mu;
  const float o = (v - mu) * rsqrtf(var + 1e-5f) * w[d] + b[d];
  dst[r * D_ + d] = o;
  if (dstb) dstb[r * D_ + d] = to_bf(o);
}

// LayerNorm for chan-mix output: h2t is [c][NROWS][PPAD] (c = feature dim).
__global__ __launch_bounds__(128)
void ln_chan_kernel(const float* __restrict__ h2t, const float* __restrict__ resx,
                    float* __restrict__ dst, bf16_t* __restrict__ dstb,
                    const float* __restrict__ w, const float* __restrict__ b)
{
  const long long r = blockIdx.x;               // r = n*PNUM_ + q
  const int c = threadIdx.x;
  const long long n = r / PNUM_;
  const int q = (int)(r % PNUM_);
  float v = h2t[(long long)c * ((long long)NROWS * PPAD) + n * PPAD + q] + resx[r * D_ + c];
  float s = v, qq = v * v;
#pragma unroll
  for (int off = 16; off > 0; off >>= 1) { s += __shfl_down(s, off, 32); qq += __shfl_down(qq, off, 32); }
  __shared__ float s1[4], s2[4];
  const int wid = c >> 5;
  if ((c & 31) == 0) { s1[wid] = s; s2[wid] = qq; }
  __syncthreads();
  s  = s1[0] + s1[1] + s1[2] + s1[3];
  qq = s2[0] + s2[1] + s2[2] + s2[3];
  const float mu  = s * (1.f / 128.f);
  const float var = qq * (1.f / 128.f) - mu * mu;
  const float o = (v - mu) * rsqrtf(var + 1e-5f) * w[c] + b[c];
  dst[r * D_ + c] = o;
  if (dstb) dstb[r * D_ + c] = to_bf(o);
}

// x[n][88][128] f32 -> xt[c][n][96] bf16 (p zero-padded 88..95)
__global__ void transpose_pad_kernel(const float* __restrict__ x, bf16_t* __restrict__ xt)
{
  const long long idx = (long long)blockIdx.x * 256 + threadIdx.x;
  if (idx >= (long long)D_ * NROWS * PPAD) return;
  const int p = (int)(idx % PPAD);
  const long long t = idx / PPAD;
  const int n = (int)(t % NROWS);
  const int c = (int)(t / NROWS);
  const float v = (p < PNUM_) ? x[((long long)n * PNUM_ + p) * D_ + c] : 0.f;
  xt[idx] = to_bf(v);
}

// xcomb = x * combW[0,0] + combB[0]  (in place)
__global__ void comb_kernel(float* __restrict__ x, const float* __restrict__ cw,
                            const float* __restrict__ cb)
{
  const long long idx = (long long)blockIdx.x * 256 + threadIdx.x;
  if (idx >= (long long)NROWS * PNUM_ * D_) return;
  x[idx] = x[idx] * cw[0] + cb[0];
}

// Circular autocorrelation alpha (replaces rfft/irfft; ortho norm -> 1/8 scale)
__global__ __launch_bounds__(128)
void scorr_kernel(const float* __restrict__ aln, const float* __restrict__ bln,
                  float* __restrict__ alpha)
{
  const int n = blockIdx.x, d = threadIdx.x;
  __shared__ float sa[POLD][D_];
  __shared__ float sb[POLD][D_];
  for (int m = 0; m < POLD; ++m) {
    sa[m][d] = aln[((long long)n * POLD + m) * D_ + d];
    sb[m][d] = bln[((long long)n * POLD + m) * D_ + d];
  }
  __syncthreads();
  float num = 0.f, den = 0.f;
  for (int j = 0; j < POLD; ++j) {
    float ra = 0.f, rb = 0.f;
    for (int m = 0; m < POLD; ++m) {
      const int mj = (m + j) & (POLD - 1);
      ra += sa[m][d] * sa[mj][d];
      rb += sb[m][d] * sb[mj][d];
    }
    ra = fmaxf(ra * 0.125f, 0.f);  // irfft(rfft_ortho*conj) = autocorr/sqrt(64)
    rb = fmaxf(rb * 0.125f, 0.f);
    num += ra * rb;
    den += rb * rb;
  }
  alpha[(long long)n * D_ + d] = sqrtf(num / (den + 1e-3f));
}

// xst[n][d][96] = bf16(alpha[n][d] * x[n][p][d]), p padded -> head GEMM A
__global__ void scale_transpose_kernel(const float* __restrict__ x,
                                       const float* __restrict__ alpha,
                                       bf16_t* __restrict__ xt)
{
  const long long idx = (long long)blockIdx.x * 256 + threadIdx.x;
  if (idx >= (long long)NROWS * D_ * PPAD) return;
  const int p = (int)(idx % PPAD);
  const long long t = idx / PPAD;
  const int d = (int)(t % D_);
  const int n = (int)(t / D_);
  const float v = (p < PNUM_)
      ? x[((long long)n * PNUM_ + p) * D_ + d] * alpha[(long long)n * D_ + d] : 0.f;
  xt[idx] = to_bf(v);
}

// De-norm + write output (B, PRED, CIN) f32
__global__ void final_kernel(const float* __restrict__ yhat, const float* __restrict__ rw,
                             const float* __restrict__ rb, const float* __restrict__ meanI,
                             const float* __restrict__ stdI, float* __restrict__ out)
{
  const int idx = blockIdx.x * 256 + threadIdx.x;
  if (idx >= B_ * PRED * CIN) return;
  const int c = idx % CIN;
  const int t = (idx / CIN) % PRED;
  const int bb = idx / (CIN * PRED);
  const int n = bb * CIN + c;
  float v = yhat[(long long)n * PRED + t];
  v = (v - rb[c]) / (rw[c] + 1e-10f) * stdI[n] + meanI[n];
  out[idx] = v;
}

// ------------------------- weight conversion ------------------------------
__global__ void conv_dense_kernel(const float* __restrict__ in, bf16_t* __restrict__ o, long long n)
{
  const long long i = (long long)blockIdx.x * 256 + threadIdx.x;
  if (i < n) o[i] = to_bf(in[i]);
}
// cW [4*D_][88][88] -> [4*D_][96][96] zero-padded
__global__ void conv_pad_cw_kernel(const float* __restrict__ in, bf16_t* __restrict__ o)
{
  const long long i = (long long)blockIdx.x * 256 + threadIdx.x;
  if (i >= 4ll * D_ * PPAD * PPAD) return;
  const int p = (int)(i % PPAD);
  const long long t = i / PPAD;
  const int q = (int)(t % PPAD);
  const long long s = t / PPAD;
  const float v = (q < PNUM_ && p < PNUM_) ? in[(s * PNUM_ + q) * PNUM_ + p] : 0.f;
  o[i] = to_bf(v);
}
// headW [192][d*88+p] -> [192][d*96+p] zero-padded
__global__ void conv_head_kernel(const float* __restrict__ in, bf16_t* __restrict__ o)
{
  const long long i = (long long)blockIdx.x * 256 + threadIdx.x;
  if (i >= (long long)PRED * KHEAD) return;
  const int p = (int)(i % PPAD);
  const long long t = i / PPAD;
  const int d = (int)(t % D_);
  const int ot = (int)(t / D_);
  const float v = (p < PNUM_) ? in[((long long)ot * D_ + d) * PNUM_ + p] : 0.f;
  o[i] = to_bf(v);
}

// ---------------------------------------------------------------------------
extern "C" void kernel_launch(void* const* d_in, const int* in_sizes, int n_in,
                              void* d_out, int out_size, void* d_ws, size_t ws_size,
                              hipStream_t stream)
{
  const float* x_input = (const float*)d_in[0];
  const float* revin_w = (const float*)d_in[1];
  const float* revin_b = (const float*)d_in[2];
  const float* predW   = (const float*)d_in[3];
  const float* predB   = (const float*)d_in[4];
  const float* embW    = (const float*)d_in[5];
  const float* tW1     = (const float*)d_in[6];
  const float* tb1     = (const float*)d_in[7];
  const float* tW2     = (const float*)d_in[8];
  const float* tb2     = (const float*)d_in[9];
  const float* tlnw    = (const float*)d_in[10];
  const float* tlnb    = (const float*)d_in[11];
  const float* cW1     = (const float*)d_in[12];
  const float* cb1     = (const float*)d_in[13];
  const float* cW2     = (const float*)d_in[14];
  const float* cb2     = (const float*)d_in[15];
  const float* clnw    = (const float*)d_in[16];
  const float* clnb    = (const float*)d_in[17];
  const float* lnw     = (const float*)d_in[18];
  const float* lnb     = (const float*)d_in[19];
  const float* combW   = (const float*)d_in[20];
  const float* combB   = (const float*)d_in[21];
  const float* headW   = (const float*)d_in[22];
  const float* headB   = (const float*)d_in[23];
  float* out = (float*)d_out;

  // Workspace carving (~460 MB total), aliased across pipeline phases.
  char* ws = (char*)d_ws;
  size_t off = 0;
  auto alloc = [&](size_t bytes) -> char* {
    char* p = ws + off;
    off = (off + bytes + 255) & ~(size_t)255;
    return p;
  };
  bf16_t* predW_bf = (bf16_t*)alloc((size_t)OUTL * SEQ * 2);
  bf16_t* tW1_bf   = (bf16_t*)alloc(4ull * PNUM_ * D_ * D_ * 2);
  bf16_t* tW2_bf   = (bf16_t*)alloc(4ull * PNUM_ * D_ * D_ * 2);
  bf16_t* cW1_bf   = (bf16_t*)alloc(4ull * D_ * PPAD * PPAD * 2);
  bf16_t* cW2_bf   = (bf16_t*)alloc(4ull * D_ * PPAD * PPAD * 2);
  bf16_t* headW_bf = (bf16_t*)alloc((size_t)PRED * KHEAD * 2);
  float*  meanB = (float*)alloc(NROWS * 4);
  float*  stdB  = (float*)alloc(NROWS * 4);
  float*  peB   = (float*)alloc((size_t)PNUM_ * D_ * 4);
  float*  alphaB= (float*)alloc((size_t)NROWS * D_ * 4);
  float*  yhatB = (float*)alloc((size_t)NROWS * PRED * 4);
  float*  XA    = (float*)alloc((size_t)NROWS * PNUM_ * D_ * 4);   // f32 activations
  bf16_t* XB    = (bf16_t*)alloc((size_t)NROWS * PNUM_ * D_ * 2);  // bf16 mirror
  bf16_t* HT    = (bf16_t*)alloc((size_t)MAXEL * 2);  // hidden bf16 (also xn early)
  float*  H2    = (float*)alloc((size_t)MAXEL * 4);   // hidden f32 (also y, bln)
  bf16_t* XT    = (bf16_t*)alloc((size_t)MAXEL * 2);  // transposed acts (also xst)
  float*  XOLD  = (float*)alloc((size_t)NROWS * POLD * D_ * 4);
  (void)ws_size; (void)in_sizes; (void)n_in; (void)out_size;

  // --- weight prep (bf16, padded) ---
  {
    long long n1 = (long long)OUTL * SEQ;
    conv_dense_kernel<<<(int)((n1 + 255) / 256), 256, 0, stream>>>(predW, predW_bf, n1);
    long long n2 = 4ll * PNUM_ * D_ * D_;
    conv_dense_kernel<<<(int)((n2 + 255) / 256), 256, 0, stream>>>(tW1, tW1_bf, n2);
    conv_dense_kernel<<<(int)((n2 + 255) / 256), 256, 0, stream>>>(tW2, tW2_bf, n2);
    long long n3 = 4ll * D_ * PPAD * PPAD;
    conv_pad_cw_kernel<<<(int)((n3 + 255) / 256), 256, 0, stream>>>(cW1, cW1_bf);
    conv_pad_cw_kernel<<<(int)((n3 + 255) / 256), 256, 0, stream>>>(cW2, cW2_bf);
    long long n4 = (long long)PRED * KHEAD;
    conv_head_kernel<<<(int)((n4 + 255) / 256), 256, 0, stream>>>(headW, headW_bf);
    pe_init_kernel<<<(PNUM_ * D_ + 255) / 256, 256, 0, stream>>>(peB);
  }

  // --- RevIN + pred linear + patch embeds ---
  revin_kernel<<<NROWS, 256, 0, stream>>>(x_input, revin_w, revin_b, meanB, stdB, HT);
  // y[2048][704] = xn @ predW^T + predB  (f32 into H2)
  gemm_bf16_wmma<0><<<dim3(16, 6, 1), 256, 0, stream>>>(
      HT, 0, SEQ, predW_bf, 0, predB, 0, OUTL,
      H2, 0, OUTL, (bf16_t*)nullptr, 0, 0, OUTL, SEQ);
  patch_embed_kernel<<<dim3(NROWS, PNUM_), 128, 0, stream>>>(
      H2, (long long)CIN * OUTL, OUTL, 1, OUTL, PNUM_, embW, peB, XA, XB);
  patch_embed_kernel<<<dim3(NROWS, POLD), 128, 0, stream>>>(
      x_input, (long long)SEQ * CIN, 1, CIN, SEQ, POLD, embW, peB, XOLD, (bf16_t*)nullptr);

  // --- mixer chain: only (s,i) = (0,1) -> (1,1) -> (1,0) is live ---
  const int chain[3] = {1, 3, 2};  // flat index s*E + i
  const long long nTr = (long long)D_ * NROWS * PPAD;
  for (int stage = 0; stage < 3; ++stage) {
    const long long bi = chain[stage];
    // temp-mix: batched over p (88 GEMMs of 2048x128x128)
    gemm_bf16_wmma<1><<<dim3(16, 1, PNUM_), 256, 0, stream>>>(
        XB, 128, PNUM_ * D_,
        tW1_bf + bi * PNUM_ * D_ * D_, (long long)D_ * D_,
        tb1 + bi * PNUM_ * D_, D_, D_,
        (float*)nullptr, 0, 0, HT, 128, PNUM_ * D_, D_, D_);
    gemm_bf16_wmma<0><<<dim3(16, 1, PNUM_), 256, 0, stream>>>(
        HT, 128, PNUM_ * D_,
        tW2_bf + bi * PNUM_ * D_ * D_, (long long)D_ * D_,
        tb2 + bi * PNUM_ * D_, D_, D_,
        H2, 128, PNUM_ * D_, (bf16_t*)nullptr, 0, 0, D_, D_);
    ln_rows_kernel<<<NROWS * PNUM_, 128, 0, stream>>>(
        H2, XA, XA, XB, tlnw + bi * D_, tlnb + bi * D_, PNUM_, PNUM_);
    // chan-mix: batched over c (128 GEMMs of 2048x96x96, zero-padded from 88)
    transpose_pad_kernel<<<(int)((nTr + 255) / 256), 256, 0, stream>>>(XA, XT);
    gemm_bf16_wmma<1><<<dim3(16, 1, D_), 256, 0, stream>>>(
        XT, (long long)NROWS * PPAD, PPAD,
        cW1_bf + bi * D_ * PPAD * PPAD, (long long)PPAD * PPAD,
        cb1 + bi * D_ * PNUM_, PNUM_, PNUM_,
        (float*)nullptr, 0, 0, HT, (long long)NROWS * PPAD, PPAD, PPAD, PPAD);
    gemm_bf16_wmma<0><<<dim3(16, 1, D_), 256, 0, stream>>>(
        HT, (long long)NROWS * PPAD, PPAD,
        cW2_bf + bi * D_ * PPAD * PPAD, (long long)PPAD * PPAD,
        cb2 + bi * D_ * PNUM_, PNUM_, PNUM_,
        H2, (long long)NROWS * PPAD, PPAD, (bf16_t*)nullptr, 0, 0, PPAD, PPAD);
    ln_chan_kernel<<<NROWS * PNUM_, 128, 0, stream>>>(
        H2, XA, XA, XB, clnw + bi * D_, clnb + bi * D_);
  }

  // --- comb, s-correction, head, de-norm ---
  {
    const long long nc = (long long)NROWS * PNUM_ * D_;
    comb_kernel<<<(int)((nc + 255) / 256), 256, 0, stream>>>(XA, combW, combB);
  }
  ln_rows_kernel<<<NROWS * POLD, 128, 0, stream>>>(
      XOLD, (const float*)nullptr, XOLD, (bf16_t*)nullptr, lnw, lnb, POLD, POLD);
  ln_rows_kernel<<<NROWS * POLD, 128, 0, stream>>>(
      XA, (const float*)nullptr, H2, (bf16_t*)nullptr, lnw, lnb, PNUM_, POLD);
  scorr_kernel<<<NROWS, 128, 0, stream>>>(XOLD, H2, alphaB);
  {
    const long long nst = (long long)NROWS * D_ * PPAD;
    scale_transpose_kernel<<<(int)((nst + 255) / 256), 256, 0, stream>>>(XA, alphaB, XT);
  }
  gemm_bf16_wmma<0><<<dim3(16, 2, 1), 256, 0, stream>>>(
      XT, 0, KHEAD, headW_bf, 0, headB, 0, PRED,
      yhatB, 0, PRED, (bf16_t*)nullptr, 0, 0, PRED, KHEAD);
  final_kernel<<<(B_ * PRED * CIN + 255) / 256, 256, 0, stream>>>(
      yhatB, revin_w, revin_b, meanB, stdB, out);
}
